// BipartiteGAT_7851200217924
// MI455X (gfx1250) — matrix operand order repaired and checked by reading, not compiled, not run on hardware
//
#include <hip/hip_runtime.h>

// ---------------------------------------------------------------------------
// Types for CDNA5 WMMA (wave32): A/B = 16 bf16 per lane, C/D = 8 f32 per lane
// ---------------------------------------------------------------------------
typedef __bf16 bf16_t;
typedef __attribute__((ext_vector_type(16))) __bf16 v16bf;
typedef __attribute__((ext_vector_type(8)))  __bf16 v8bf;
typedef __attribute__((ext_vector_type(4)))  __bf16 v4bf;
typedef __attribute__((ext_vector_type(8)))  float  v8f;

__device__ __forceinline__ float wave_sum(float v) {
  #pragma unroll
  for (int o = 16; o > 0; o >>= 1) v += __shfl_xor(v, o, 32);
  return v;
}

// ---------------------------------------------------------------------------
// Kernel 0: f32 -> bf16 bulk convert (native v_cvt, 4 elements/thread)
// ---------------------------------------------------------------------------
__global__ void __launch_bounds__(256)
cvt_bf16_kernel(const float* __restrict__ in, bf16_t* __restrict__ out, int n) {
  int i = (blockIdx.x * blockDim.x + threadIdx.x) * 4;
  if (i >= n) return;
  float4 f = *(const float4*)(in + i);
  v4bf o;
  o[0] = (bf16_t)f.x; o[1] = (bf16_t)f.y; o[2] = (bf16_t)f.z; o[3] = (bf16_t)f.w;
  *(v4bf*)(out + i) = o;
}

// ---------------------------------------------------------------------------
// Kernel 1: fold attention vectors through W.
//   u[h,c] = sum_d a[h,d]    * W[h*32+d, c]   (et = feat . u[h])
//   v[h,c] = sum_d a[h,32+d] * W[h*32+d, c]   (es = feat . v[h])
// ---------------------------------------------------------------------------
__global__ void fold_kernel(const float* __restrict__ W, const float* __restrict__ a,
                            float* __restrict__ u, float* __restrict__ v) {
  int idx = blockIdx.x * blockDim.x + threadIdx.x;   // h*256 + c, total 2048
  if (idx >= 8 * 256) return;
  int h = idx >> 8, c = idx & 255;
  float su = 0.f, sv = 0.f;
  #pragma unroll 4
  for (int d = 0; d < 32; ++d) {
    float w = W[(size_t)(h * 32 + d) * 256 + c];
    su += a[h * 64 + d] * w;
    sv += a[h * 64 + 32 + d] * w;
  }
  u[idx] = su;
  v[idx] = sv;
}

// ---------------------------------------------------------------------------
// Kernel 2: out[h*nrows + n] = dot(feat[n, 0:256], vecs[h, 0:256])
// One wave per (head, row); block = 256 threads = 8 waves = 8 heads.
// ---------------------------------------------------------------------------
__global__ void __launch_bounds__(256)
dotvec_kernel(const float* __restrict__ feat, const float* __restrict__ vecs,
              float* __restrict__ out, int nrows) {
  int n = blockIdx.x;
  int h = threadIdx.x >> 5;
  int lane = threadIdx.x & 31;
  const float* fr = feat + (size_t)n * 256;
  const float* vr = vecs + (size_t)h * 256;
  float s = 0.f;
  #pragma unroll
  for (int k = lane; k < 256; k += 32) s += fr[k] * vr[k];
  s = wave_sum(s);
  if (lane == 0) out[(size_t)h * nrows + n] = s;
}

// ---------------------------------------------------------------------------
// Kernel 3: S_T[c, m] = sum_k W[c,k] * feat[m,k]   (transposed source transform)
// A = W tile (16 c-rows x 32 k), B = feat^T (32 k x 16 m).  All inputs bf16.
// Block = 128 threads (4 waves); tile = 64 c-rows x 32 m-cols; K loop over 256.
// Every fragment is a contiguous LDS read (v8bf/v16bf) -> ds_load_b128s.
// ---------------------------------------------------------------------------
__global__ void __launch_bounds__(128)
gemm_st_kernel(const bf16_t* __restrict__ Wb,    // [256, 256] bf16
               const bf16_t* __restrict__ featb, // [R, 256] bf16
               bf16_t* __restrict__ S_T,         // [256, Ms] bf16
               int Ms) {
  __shared__ __align__(32) bf16_t Wlds[4][16 * 32];   // per-wave [c][k]
  __shared__ __align__(32) bf16_t Flds[32 * 32];      // [m][k]
  const int tid = threadIdx.x;
  const int w = tid >> 5, lane = tid & 31;
  const int c0 = blockIdx.x * 64;
  const int m0 = blockIdx.y * 32;
  const int hl = lane >> 4;            // K-half selector (lanes 16-31)
  const int rr = lane & 15;            // M row / N col within fragment
  v8f acc0 = {0.f,0.f,0.f,0.f,0.f,0.f,0.f,0.f};
  v8f acc1 = {0.f,0.f,0.f,0.f,0.f,0.f,0.f,0.f};
  for (int k0 = 0; k0 < 256; k0 += 32) {
    {   // feat tile: 32 m-rows x 32 k (64B/row), one uint4 per thread
      int mi = tid >> 2, q = tid & 3;
      ((uint4*)Flds)[mi * 4 + q] =
          ((const uint4*)(featb + (size_t)(m0 + mi) * 256 + k0))[q];
    }
    {   // W tile for wave w: 16 c-rows x 32 k, two uint4 per lane
      #pragma unroll
      for (int i = lane; i < 64; i += 32) {
        int c = i >> 2, q = i & 3;
        ((uint4*)Wlds[w])[i] =
            ((const uint4*)(Wb + (size_t)(c0 + 16 * w + c) * 256 + k0))[q];
      }
    }
    __syncthreads();
    // A fragment: elements 0-15 -> K = hl*16 .. +15 (contiguous)
    v8bf alo = *(const v8bf*)&Wlds[w][rr * 32 + hl * 16];
    v8bf ahi = *(const v8bf*)&Wlds[w][rr * 32 + hl * 16 + 8];
    v16bf af = __builtin_shufflevector(alo, ahi, 0,1,2,3,4,5,6,7,
                                                 8,9,10,11,12,13,14,15);
    // B fragments: element e -> K = hl*16 + e (contiguous 32B)
    v16bf b0 = *(const v16bf*)&Flds[rr * 32 + hl * 16];
    v16bf b1 = *(const v16bf*)&Flds[(rr + 16) * 32 + hl * 16];
    acc0 = __builtin_amdgcn_wmma_f32_16x16x32_bf16(false, af, false, b0,
                                                   (short)0, acc0, false, false);
    acc1 = __builtin_amdgcn_wmma_f32_16x16x32_bf16(false, af, false, b1,
                                                   (short)0, acc1, false, false);
    __syncthreads();
  }
  #pragma unroll
  for (int r = 0; r < 8; ++r) {        // D layout: VGPR r -> M = r + hl*8, N = rr
    int c = c0 + 16 * w + r + hl * 8;
    S_T[(size_t)c * Ms + m0 + rr]      = (bf16_t)acc0[r];
    S_T[(size_t)c * Ms + m0 + 16 + rr] = (bf16_t)acc1[r];
  }
}

// ---------------------------------------------------------------------------
// Kernel 4: per-(head,row) softmax stats over masked leaky(et+es).
// One block per target row; adj row staged in LDS; wave h = head h; online
// (max, sum-of-exp) with wave-level combine.  adj(n,m) = adj[n*rs + m*cs].
// ---------------------------------------------------------------------------
__global__ void __launch_bounds__(256)
softmax_stats_kernel(const int* __restrict__ adj,
                     const float* __restrict__ et, const float* __restrict__ es,
                     float* __restrict__ rmax, float* __restrict__ rsum,
                     int Nt, int Ms, long rs, long cs) {
  __shared__ int amask[4096];
  const int n = blockIdx.x;
  for (int m = threadIdx.x; m < Ms; m += blockDim.x)
    amask[m] = adj[(size_t)n * rs + (size_t)m * cs];
  __syncthreads();
  const int h = threadIdx.x >> 5;
  const int lane = threadIdx.x & 31;
  const float etv = et[(size_t)h * Nt + n];
  float mx = -3.0e38f, sum = 0.f;
  for (int m = lane; m < Ms; m += 32) {
    float e;
    if (amask[m] > 0) {
      e = etv + es[(size_t)h * Ms + m];
      e = (e >= 0.f) ? e : 0.2f * e;     // LeakyReLU(0.2)
    } else {
      e = -9.0e15f;                      // NEG mask
    }
    float nm = fmaxf(mx, e);
    sum = sum * __expf(mx - nm) + __expf(e - nm);
    mx = nm;
  }
  #pragma unroll
  for (int o = 16; o > 0; o >>= 1) {
    float omx = __shfl_xor(mx, o, 32);
    float osm = __shfl_xor(sum, o, 32);
    float nm = fmaxf(mx, omx);
    sum = sum * __expf(mx - nm) + osm * __expf(omx - nm);
    mx = nm;
  }
  if (lane == 0) {
    rmax[(size_t)h * Nt + n] = mx;
    rsum[(size_t)h * Nt + n] = sum;
  }
}

// ---------------------------------------------------------------------------
// Kernel 5: fused attention aggregation + BN(eval) + ReLU.
// Grid: (Nt/64, 8 heads). Block = 128 threads (4 waves); wave owns a 16-row
// strip of the 64x32 (rows x head-cols) output tile.  Per 32-wide source chunk:
// A-frag = exp(leaky(et+es)-rowmax)*mask built in registers (native bf16 cvt),
// B-frags = contiguous v16bf LDS reads of the S_T tile, 2x WMMA.
// Epilogue: /rowsum, BN, ReLU, f32 store.
// ---------------------------------------------------------------------------
__global__ void __launch_bounds__(128)
gat_aggregate_kernel(const int* __restrict__ adj, long rs, long cs,
                     const float* __restrict__ et, const float* __restrict__ es,
                     const float* __restrict__ rmax, const float* __restrict__ rsum,
                     const bf16_t* __restrict__ S_T,   // [256, Ms]
                     const float* __restrict__ gamma, const float* __restrict__ beta,
                     float* __restrict__ out, int Nt, int Ms) {
  __shared__ int a_tile[4][16 * 32];                 // per-wave adj block
  __shared__ __align__(32) bf16_t st[32 * 32];       // [c][k] S_T chunk
  __shared__ float es_s[32];
  const int tid = threadIdx.x;
  const int w = tid >> 5, lane = tid & 31;
  const int h = blockIdx.y;
  const int n0 = blockIdx.x * 64;
  const int hl = lane >> 4;
  const int rr = lane & 15;
  const int c0 = h * 32;
  const int nrow = n0 + 16 * w + rr;                 // A-fragment row
  const float etv = et[(size_t)h * Nt + nrow];
  const float rmx = rmax[(size_t)h * Nt + nrow];
  v8f acc0 = {0.f,0.f,0.f,0.f,0.f,0.f,0.f,0.f};
  v8f acc1 = {0.f,0.f,0.f,0.f,0.f,0.f,0.f,0.f};
  for (int m0 = 0; m0 < Ms; m0 += 32) {
    {   // S_T tile: 32 c-rows x 32 m (64B/row), one uint4 per thread
      int ci = tid >> 2, q = tid & 3;
      ((uint4*)st)[ci * 4 + q] =
          ((const uint4*)(S_T + (size_t)(c0 + ci) * Ms + m0))[q];
      if (m0 + 32 < Ms)   // prefetch next chunk (global_prefetch_b8)
        __builtin_prefetch(S_T + (size_t)(c0 + ci) * Ms + m0 + 32, 0, 1);
    }
    if (tid < 32) es_s[tid] = es[(size_t)h * Ms + m0 + tid];
    for (int i = lane; i < 512; i += 32) {           // wave's 16x32 adj block
      int r = i >> 5, mm = i & 31;
      a_tile[w][i] = adj[(size_t)(n0 + 16 * w + r) * rs + (size_t)(m0 + mm) * cs];
    }
    __syncthreads();
    v16bf af, b0, b1;
    #pragma unroll
    for (int e = 0; e < 16; ++e) {
      int k = e + hl * 16;                           // A 16x32 K map
      float ev = etv + es_s[k];
      ev = (ev >= 0.f) ? ev : 0.2f * ev;
      float p = (a_tile[w][rr * 32 + k] > 0) ? __expf(ev - rmx) : 0.f;
      af[e] = (bf16_t)p;                             // native cvt
    }
    b0 = *(const v16bf*)&st[rr * 32 + hl * 16];
    b1 = *(const v16bf*)&st[(rr + 16) * 32 + hl * 16];
    acc0 = __builtin_amdgcn_wmma_f32_16x16x32_bf16(false, af, false, b0,
                                                   (short)0, acc0, false, false);
    acc1 = __builtin_amdgcn_wmma_f32_16x16x32_bf16(false, af, false, b1,
                                                   (short)0, acc1, false, false);
    __syncthreads();
  }
  const float bnscale = rsqrtf(1.0f + 1e-5f);        // running_var=1, eps
  const float g0 = gamma[c0 + rr] * bnscale,      bt0 = beta[c0 + rr];
  const float g1 = gamma[c0 + rr + 16] * bnscale, bt1 = beta[c0 + rr + 16];
  #pragma unroll
  for (int r = 0; r < 8; ++r) {
    int n = n0 + 16 * w + r + hl * 8;
    float inv = 1.0f / rsum[(size_t)h * Nt + n];
    float v0 = fmaf(acc0[r] * inv, g0, bt0);  v0 = v0 > 0.f ? v0 : 0.f;
    float v1 = fmaf(acc1[r] * inv, g1, bt1);  v1 = v1 > 0.f ? v1 : 0.f;
    out[(size_t)n * 256 + c0 + rr]      = v0;
    out[(size_t)n * 256 + c0 + rr + 16] = v1;
  }
}

// ---------------------------------------------------------------------------
// Host launcher
// ---------------------------------------------------------------------------
extern "C" void kernel_launch(void* const* d_in, const int* in_sizes, int n_in,
                              void* d_out, int out_size, void* d_ws, size_t ws_size,
                              hipStream_t stream) {
  (void)in_sizes; (void)n_in; (void)out_size; (void)ws_size;
  const float* c_feat = (const float*)d_in[0];   // [4096, 256]
  const float* d_feat = (const float*)d_in[1];   // [2048, 256]
  const int*   adj    = (const int*)  d_in[2];   // [4096, 2048]
  const float* c_W    = (const float*)d_in[3];   // [256, 256]
  const float* c_a    = (const float*)d_in[4];   // [8, 64]
  const float* d_W    = (const float*)d_in[5];
  const float* d_a    = (const float*)d_in[6];
  const float* c_g    = (const float*)d_in[7];
  const float* c_b    = (const float*)d_in[8];
  const float* d_g    = (const float*)d_in[9];
  const float* d_b    = (const float*)d_in[10];
  float* out = (float*)d_out;

  constexpr int N = 4096, M = 2048, HD = 256, H = 8;

  char* ws = (char*)d_ws;
  size_t off = 0;
  auto alloc_f = [&](size_t n) { float* p = (float*)(ws + off); off += n * sizeof(float); return p; };
  auto alloc_b = [&](size_t n) { bf16_t* p = (bf16_t*)(ws + off); off += n * sizeof(bf16_t); return p; };
  float* u_c    = alloc_f((size_t)H * HD);
  float* v_c    = alloc_f((size_t)H * HD);
  float* u_d    = alloc_f((size_t)H * HD);
  float* v_d    = alloc_f((size_t)H * HD);
  float* et_c   = alloc_f((size_t)H * N);
  float* es_c   = alloc_f((size_t)H * M);
  float* et_d   = alloc_f((size_t)H * M);
  float* es_d   = alloc_f((size_t)H * N);
  float* rmax_c = alloc_f((size_t)H * N);
  float* rsum_c = alloc_f((size_t)H * N);
  float* rmax_d = alloc_f((size_t)H * M);
  float* rsum_d = alloc_f((size_t)H * M);
  bf16_t* cfb  = alloc_b((size_t)N * HD);   // bf16 copies of inputs
  bf16_t* dfb  = alloc_b((size_t)M * HD);
  bf16_t* cWb  = alloc_b((size_t)HD * HD);
  bf16_t* dWb  = alloc_b((size_t)HD * HD);
  bf16_t* ST_c = alloc_b((size_t)M * HD);   // S_c^T : [256, 2048]
  bf16_t* ST_d = alloc_b((size_t)N * HD);   // S_d^T : [256, 4096]

  // 0) bulk f32 -> bf16 (native v_cvt)
  cvt_bf16_kernel<<<(N * HD / 4 + 255) / 256, 256, 0, stream>>>(c_feat, cfb, N * HD);
  cvt_bf16_kernel<<<(M * HD / 4 + 255) / 256, 256, 0, stream>>>(d_feat, dfb, M * HD);
  cvt_bf16_kernel<<<(HD * HD / 4 + 255) / 256, 256, 0, stream>>>(c_W, cWb, HD * HD);
  cvt_bf16_kernel<<<(HD * HD / 4 + 255) / 256, 256, 0, stream>>>(d_W, dWb, HD * HD);

  // 1) fold a through W (tiny)
  fold_kernel<<<8, 256, 0, stream>>>(c_W, c_a, u_c, v_c);
  fold_kernel<<<8, 256, 0, stream>>>(d_W, d_a, u_d, v_d);

  // 2) et/es logits (wave-per-row dots, f32)
  dotvec_kernel<<<N, 256, 0, stream>>>(c_feat, u_c, et_c, N);   // et_c[h,n]
  dotvec_kernel<<<M, 256, 0, stream>>>(d_feat, v_c, es_c, M);   // es_c[h,m]
  dotvec_kernel<<<M, 256, 0, stream>>>(d_feat, u_d, et_d, M);   // et_d[h,n]
  dotvec_kernel<<<N, 256, 0, stream>>>(c_feat, v_d, es_d, N);   // es_d[h,m]

  // 3) transposed source transforms S_T = W @ feat^T (WMMA bf16)
  gemm_st_kernel<<<dim3(HD / 64, M / 32), 128, 0, stream>>>(cWb, dfb, ST_c, M);
  gemm_st_kernel<<<dim3(HD / 64, N / 32), 128, 0, stream>>>(dWb, cfb, ST_d, N);

  // 4) softmax stats.  c-dir: adj[n,m] = adj[n*M + m];  d-dir: adj[m,n] = adj[n + m*M]
  softmax_stats_kernel<<<N, 256, 0, stream>>>(adj, et_c, es_c, rmax_c, rsum_c,
                                              N, M, (long)M, 1L);
  softmax_stats_kernel<<<M, 256, 0, stream>>>(adj, et_d, es_d, rmax_d, rsum_d,
                                              M, N, 1L, (long)M);

  // 5) fused attention @ S + BN + ReLU (WMMA bf16)
  gat_aggregate_kernel<<<dim3(N / 64, H), 128, 0, stream>>>(
      adj, (long)M, 1L, et_c, es_c, rmax_c, rsum_c, ST_c, c_g, c_b,
      out, N, M);
  gat_aggregate_kernel<<<dim3(M / 64, H), 128, 0, stream>>>(
      adj, 1L, (long)M, et_d, es_d, rmax_d, rsum_d, ST_d, d_g, d_b,
      out + (size_t)N * HD, M, N);
}
